// SpikeFP16Linear_MultiPrecision_76312978916071
// MI455X (gfx1250) — matrix operand order repaired and checked by reading, not compiled, not run on hardware
//
#include <hip/hip_runtime.h>

typedef _Float16 half8  __attribute__((ext_vector_type(8)));
typedef _Float16 half16 __attribute__((ext_vector_type(16)));
typedef float    f32x4  __attribute__((ext_vector_type(4)));
typedef float    f32x8  __attribute__((ext_vector_type(8)));

static constexpr int kB   = 4096;   // batch rows (M)
static constexpr int kIn  = 1024;   // reduction  (K)
static constexpr int kOut = 4096;   // out features (N)

// ---------------------------------------------------------------------------
// Pass 1: pulse[...,16] (float32 0/1) -> packed fp16 value.
// Pure streaming: 64 B read -> 2 B write per element. NT loads keep the
// 512 MB pulse stream out of L2 so the decoded 16 MB matrices stay resident.
// ---------------------------------------------------------------------------
__global__ __launch_bounds__(256)
void decode_pulse_kernel(const float* __restrict__ pulse,
                         _Float16* __restrict__ vals, int n) {
  int idx = blockIdx.x * blockDim.x + threadIdx.x;
  if (idx >= n) return;
  const f32x4* p = reinterpret_cast<const f32x4*>(pulse) + (size_t)idx * 4;
  f32x4 q0 = __builtin_nontemporal_load(p + 0);
  f32x4 q1 = __builtin_nontemporal_load(p + 1);
  f32x4 q2 = __builtin_nontemporal_load(p + 2);
  f32x4 q3 = __builtin_nontemporal_load(p + 3);
  float v[16] = {q0[0], q0[1], q0[2], q0[3], q1[0], q1[1], q1[2], q1[3],
                 q2[0], q2[1], q2[2], q2[3], q3[0], q3[1], q3[2], q3[3]};
  unsigned b = 0;
#pragma unroll
  for (int j = 0; j < 16; ++j) b |= (v[j] != 0.0f) ? (1u << (15 - j)) : 0u;
  vals[idx] = __builtin_bit_cast(_Float16, (unsigned short)b);
}

// ---------------------------------------------------------------------------
// Pass 2: C = X * W^T via v_wmma_f32_16x16x32_f16, fused fp16-round +
// pulse-expansion epilogue with non-temporal b128 stores.
// Wave tile 32x64 (2x4 fragments), 8 waves -> 128x128 workgroup tile.
// X,W are 8 MB each -> fully L2 resident; no LDS staging needed.
// ---------------------------------------------------------------------------
__global__ __launch_bounds__(256)
void wmma_gemm_pulse_kernel(const _Float16* __restrict__ X,
                            const _Float16* __restrict__ W,
                            float* __restrict__ outPulse) {
  const int lane    = threadIdx.x & 31;
  const int wave    = threadIdx.x >> 5;
  const int lane15  = lane & 15;
  const int halfSel = lane >> 4;               // 0: lanes 0-15, 1: lanes 16-31

  const int m0 = blockIdx.y * 128 + (wave & 3) * 32;   // 2 A-fragments of 16
  const int n0 = blockIdx.x * 128 + (wave >> 2) * 64;  // 4 B-fragments of 16

  // A fragment (16x32 f16): lane<16 holds row K=0..7 (v0-3) and K=16..23
  // (v4-7); lanes 16-31 shift K by +8.
  const _Float16* aPtr[2];
#pragma unroll
  for (int am = 0; am < 2; ++am)
    aPtr[am] = X + (size_t)(m0 + am * 16 + lane15) * kIn + halfSel * 8;

  // B fragment (32x16 f16): lane<16 holds col K=0..7 and K=16..23; lanes 16-31 shift K by +8.
  const _Float16* bPtr[4];
#pragma unroll
  for (int bn = 0; bn < 4; ++bn)
    bPtr[bn] = W + (size_t)(n0 + bn * 16 + lane15) * kIn + halfSel * 8;

  f32x8 acc[2][4] = {};

  for (int k0 = 0; k0 < kIn; k0 += 32) {
    half16 afr[2], bfr[4];
#pragma unroll
    for (int am = 0; am < 2; ++am) {
      half8 lo = *(const half8*)(aPtr[am] + k0);
      half8 hi = *(const half8*)(aPtr[am] + k0 + 16);
      afr[am] = __builtin_shufflevector(lo, hi, 0, 1, 2, 3, 4, 5, 6, 7,
                                        8, 9, 10, 11, 12, 13, 14, 15);
    }
#pragma unroll
    for (int bn = 0; bn < 4; ++bn) {
      half8 lo = *(const half8*)(bPtr[bn] + k0);
      half8 hi = *(const half8*)(bPtr[bn] + k0 + 16);
      bfr[bn] = __builtin_shufflevector(lo, hi, 0, 1, 2, 3, 4, 5, 6, 7,
                                        8, 9, 10, 11, 12, 13, 14, 15);
    }
#pragma unroll
    for (int am = 0; am < 2; ++am)
#pragma unroll
      for (int bn = 0; bn < 4; ++bn)
        acc[am][bn] = __builtin_amdgcn_wmma_f32_16x16x32_f16(
            /*neg_a=*/false, afr[am], /*neg_b=*/false, bfr[bn],
            /*c_mod=*/(short)0, acc[am][bn],
            /*reuse_a=*/false, /*reuse_b=*/false);
  }

  // Epilogue: C layout -> VGPR i holds M = i + 8*halfSel, N = lane15.
  // fp32 -> fp16 (RTE) -> 16 pulse floats, 4x b128 NT stores per element.
#pragma unroll
  for (int am = 0; am < 2; ++am) {
#pragma unroll
    for (int bn = 0; bn < 4; ++bn) {
      const int n = n0 + bn * 16 + lane15;
#pragma unroll
      for (int i = 0; i < 8; ++i) {
        const int m = m0 + am * 16 + halfSel * 8 + i;
        const unsigned short bits =
            __builtin_bit_cast(unsigned short, (_Float16)acc[am][bn][i]);
        f32x4* o =
            reinterpret_cast<f32x4*>(outPulse + ((size_t)m * kOut + n) * 16);
#pragma unroll
        for (int q = 0; q < 4; ++q) {
          f32x4 wv;
          wv[0] = (float)((bits >> (15 - (q * 4 + 0))) & 1);
          wv[1] = (float)((bits >> (15 - (q * 4 + 1))) & 1);
          wv[2] = (float)((bits >> (15 - (q * 4 + 2))) & 1);
          wv[3] = (float)((bits >> (15 - (q * 4 + 3))) & 1);
          __builtin_nontemporal_store(wv, o + q);
        }
      }
    }
  }
}

extern "C" void kernel_launch(void* const* d_in, const int* in_sizes, int n_in,
                              void* d_out, int out_size, void* d_ws,
                              size_t ws_size, hipStream_t stream) {
  (void)in_sizes; (void)n_in; (void)out_size; (void)ws_size;
  const float* x_pulse = (const float*)d_in[0];   // [4096, 1024, 16]
  const float* w_pulse = (const float*)d_in[1];   // [4096, 1024, 16]
  float* out = (float*)d_out;                     // [4096, 4096, 16]

  _Float16* xh = (_Float16*)d_ws;                 // 8 MB
  _Float16* wh = xh + (size_t)kB * kIn;           // 8 MB

  const int nx = kB * kIn;
  const int nw = kOut * kIn;
  decode_pulse_kernel<<<(nx + 255) / 256, 256, 0, stream>>>(x_pulse, xh, nx);
  decode_pulse_kernel<<<(nw + 255) / 256, 256, 0, stream>>>(w_pulse, wh, nw);

  dim3 grid(kOut / 128, kB / 128);
  wmma_gemm_pulse_kernel<<<grid, 256, 0, stream>>>(xh, wh, out);
}
